// FlashAttention_67095979099032
// MI455X (gfx1250) — matrix operand order repaired and checked by reading, not compiled, not run on hardware
//
#include <hip/hip_runtime.h>
#include <hip/hip_bf16.h>

typedef __bf16 bf16_t;
typedef __attribute__((ext_vector_type(8)))  __bf16 v8bf;
typedef __attribute__((ext_vector_type(16))) __bf16 v16bf;
typedef __attribute__((ext_vector_type(8)))  float  v8f;
typedef __attribute__((ext_vector_type(4)))  unsigned int u32x4;
typedef __attribute__((ext_vector_type(8)))  int i32x8;
typedef __attribute__((ext_vector_type(4)))  int i32x4;

union V16 { v16bf v; v8bf h[2]; };

__device__ inline v8f wmma_bf16(v16bf a, v16bf b, v8f c) {
    return __builtin_amdgcn_wmma_f32_16x16x32_bf16(false, a, false, b, (short)0, c, false, false);
}

#if __has_builtin(__builtin_amdgcn_tensor_load_to_lds)
#define USE_TDM 1
#else
#define USE_TDM 0
#endif

__device__ inline void wait_tensorcnt0() {
#if __has_builtin(__builtin_amdgcn_s_wait_tensorcnt)
    __builtin_amdgcn_s_wait_tensorcnt(0);
#else
    asm volatile("s_wait_tensorcnt 0x0" ::: "memory");
#endif
}

#if USE_TDM
// 2-D tile DMA: global (row-major, 2-byte elems) -> LDS, per D# layout in
// CDNA5 ISA.  tile_d0 = contiguous dim.
// This toolchain exposes the 6-arg builtin:
//   (u32x4 g0, i32x8 g1, i32x4 g2, i32x4 g3, i32x8 g4, i32 cpol)
__device__ inline void tdm_load_tile_2d(unsigned lds_off, const void* gptr,
                                        unsigned tensor_d0, unsigned tensor_d1,
                                        unsigned tile_d0, unsigned tile_d1,
                                        unsigned stride0_elems) {
    unsigned long long ga = (unsigned long long)gptr;
    u32x4 g0;
    g0[0] = 1u;                                    // count=1, user mode
    g0[1] = lds_off;                               // lds_addr (bytes)
    g0[2] = (unsigned)ga;                          // global_addr[31:0]
    g0[3] = (unsigned)(ga >> 32) | (2u << 30);     // global_addr[56:32] | type=2
    i32x8 g1;
    g1[0] = 0x00010000;                            // data_size=1 (2 bytes), wg_mask=0
    g1[1] = (int)((tensor_d0 & 0xFFFFu) << 16);    // tensor_dim0[15:0]
    g1[2] = (int)((tensor_d0 >> 16) | ((tensor_d1 & 0xFFFFu) << 16));
    g1[3] = (int)((tensor_d1 >> 16) | (tile_d0 << 16));   // tile_dim0
    g1[4] = (int)(tile_d1 & 0xFFFFu);              // tile_dim1 (tile_dim2=0)
    g1[5] = (int)stride0_elems;                    // tensor_dim0_stride[31:0]
    g1[6] = 0;                                     // stride0 hi / dim1_stride lo
    g1[7] = 0;
    i32x4 g2 = {0, 0, 0, 0};
    i32x4 g3 = {0, 0, 0, 0};
    i32x8 g4 = {0, 0, 0, 0, 0, 0, 0, 0};
    __builtin_amdgcn_tensor_load_to_lds(g0, g1, g2, g3, g4, 0);
}
// Flat LDS pointers map to LDS via addr[31:0] (ISA §10.2 aperture table).
__device__ inline unsigned lds_off_of(const void* p) {
    return (unsigned)(unsigned long long)p;
}
#endif

// ---------------------------------------------------------------- constants
#define HIDDEN   2048
#define NHEADS   16
#define NKVHEADS 4
#define HEADDIM  128
#define WINDOW   1024
#define BATCH    2
#define SEQ      2048
#define LOG2_THETA 13.287712379549449f   // log2(10000)

// ---------------------------------------------------------------- fp32 -> bf16
__global__ __launch_bounds__(256) void f32_to_bf16(const float* __restrict__ src,
                                                   bf16_t* __restrict__ dst, long n) {
    long i = (long)blockIdx.x * blockDim.x + threadIdx.x;
    if (i < n) dst[i] = (bf16_t)src[i];
}

// ---------------------------------------------------------------- bf16 GEMM
// C[M,N] = A[M,K] * B[K,N] + bias.  M,N,K multiples of 128.
// 256 threads (8 waves). Block tile 128x128, K step 32, double-buffered LDS.
// A tile staged by the Tensor Data Mover (wave 0 issues, s_wait_tensorcnt),
// B tile staged transposed by all threads (feeds the B-fragment layout).
__global__ __launch_bounds__(256) void gemm_bf16(const bf16_t* __restrict__ A,
                                                 const bf16_t* __restrict__ B,
                                                 const float* __restrict__ bias,
                                                 float* __restrict__ C,
                                                 int M, int N, int K) {
    __shared__ __attribute__((aligned(16))) bf16_t sA[2][128 * 32];   // [row][k]
    __shared__ __attribute__((aligned(16))) bf16_t sBT[2][128 * 32];  // [col][k]

    const int tid  = threadIdx.x;
    const int lane = tid & 31;
    const int wave = tid >> 5;
    const int lm   = lane & 15;
    const int hs   = lane >> 4;
    const int wm   = wave >> 2;       // 0..1
    const int wn   = wave & 3;        // 0..3
    const int blockN = blockIdx.x * 128;
    const int blockM = blockIdx.y * 128;

    v8f acc[4][2];
    for (int mf = 0; mf < 4; ++mf)
        for (int nf = 0; nf < 2; ++nf)
            for (int r = 0; r < 8; ++r) acc[mf][nf][r] = 0.0f;

    const int arow = tid >> 1, acol = (tid & 1) * 16;  // A tile fallback mapping
    const int brow = tid >> 3, bcol = (tid & 7) * 16;  // B tile loader mapping

    auto stage_tiles = [&](int kt, int which) {
#if USE_TDM
        if (wave == 0) {
            tdm_load_tile_2d(lds_off_of(&sA[which][0]),
                             A + (size_t)blockM * K + kt,
                             (unsigned)K, (unsigned)M, 32u, 128u, (unsigned)K);
        }
#else
        {
            const bf16_t* gp = A + (size_t)(blockM + arow) * K + kt + acol;
            *(v8bf*)(&sA[which][arow * 32 + acol])     = *(const v8bf*)gp;
            *(v8bf*)(&sA[which][arow * 32 + acol + 8]) = *(const v8bf*)(gp + 8);
        }
#endif
        {
            const bf16_t* gp = B + (size_t)(kt + brow) * N + blockN + bcol;
            union { v8bf v[2]; __bf16 e[16]; } t;
            t.v[0] = *(const v8bf*)gp;
            t.v[1] = *(const v8bf*)(gp + 8);
            for (int i = 0; i < 16; ++i) sBT[which][(bcol + i) * 32 + brow] = t.e[i];
        }
    };

    // prologue: stage first tile
    stage_tiles(0, 0);
#if USE_TDM
    if (wave == 0) wait_tensorcnt0();
#endif
    __syncthreads();

    const int nsteps = K >> 5;
    for (int i = 0; i < nsteps; ++i) {
        const int buf = i & 1;
        if (i + 1 < nsteps) stage_tiles((i + 1) << 5, buf ^ 1);

        V16 afr[4], bfr[2];
        for (int mf = 0; mf < 4; ++mf) {
            const bf16_t* p = &sA[buf][(wm * 64 + mf * 16 + lm) * 32 + hs * 8];
            afr[mf].h[0] = *(const v8bf*)p;
            afr[mf].h[1] = *(const v8bf*)(p + 16);
        }
        for (int nf = 0; nf < 2; ++nf) {
            const bf16_t* p = &sBT[buf][(wn * 32 + nf * 16 + lm) * 32 + hs * 8];
            bfr[nf].h[0] = *(const v8bf*)p;
            bfr[nf].h[1] = *(const v8bf*)(p + 16);
        }
        for (int mf = 0; mf < 4; ++mf)
            for (int nf = 0; nf < 2; ++nf)
                acc[mf][nf] = wmma_bf16(afr[mf].v, bfr[nf].v, acc[mf][nf]);

#if USE_TDM
        if (wave == 0) wait_tensorcnt0();   // next A tile landed
#endif
        __syncthreads();                    // next B tile visible, buf reusable
    }

    // epilogue: hoisted bias (uniform branch, two loads per wave)
    float bcol2[2];
    if (bias) {
        bcol2[0] = bias[blockN + wn * 32 + lm];
        bcol2[1] = bias[blockN + wn * 32 + 16 + lm];
    } else {
        bcol2[0] = 0.0f; bcol2[1] = 0.0f;
    }
    for (int mf = 0; mf < 4; ++mf)
        for (int nf = 0; nf < 2; ++nf)
            for (int r = 0; r < 8; ++r) {
                int row = blockM + wm * 64 + mf * 16 + r + hs * 8;
                int col = blockN + wn * 32 + nf * 16 + lm;
                C[(size_t)row * N + col] = acc[mf][nf][r] + bcol2[nf];
            }
}

// ---------------------------------------------------------------- RoPE + relayout
// src: [B, S, H*128] f32  ->  dst: [B, H, S, 128] bf16 (rotated if rotate!=0)
__global__ __launch_bounds__(256) void rope_relayout(const float* __restrict__ src,
                                                     bf16_t* __restrict__ dst,
                                                     const long long* __restrict__ pos,
                                                     int H, int rotate, long n) {
    long i = (long)blockIdx.x * blockDim.x + threadIdx.x;
    if (i >= n) return;
    int d = (int)(i & 127);
    long t = i >> 7;
    int s = (int)(t % SEQ); t /= SEQ;
    int h = (int)(t % H);
    int b = (int)(t / H);
    const float* row = src + ((size_t)(b * SEQ + s)) * (H * 128) + h * 128;
    float x = row[d];
    float y = x;
    if (rotate) {
        float p = (float)pos[b * SEQ + s];
        int i2 = d & 63;
        float inv = exp2f(-(2.0f * (float)i2 / 128.0f) * LOG2_THETA);
        float ang = p * inv;
        float c = __cosf(ang), sn = __sinf(ang);
        float other = (d < 64) ? row[d + 64] : row[d - 64];
        y = (d < 64) ? (x * c - other * sn) : (x * c + other * sn);
    }
    dst[i] = (bf16_t)y;
}

// ---------------------------------------------------------------- flash attention
// Q: [B,H,S,128] bf16, K/V: [B,KVH,S,128] bf16, O: [B,S,H*128] bf16
// block = 128 threads (4 waves); block handles one (b,h,64-query tile),
// each wave owns 16 query rows. Keys in blocks of 32 (sliding window 1024).
__global__ __launch_bounds__(128) void attn_kernel(const bf16_t* __restrict__ Q,
                                                   const bf16_t* __restrict__ Km,
                                                   const bf16_t* __restrict__ Vm,
                                                   bf16_t* __restrict__ O) {
    __shared__ __attribute__((aligned(16))) bf16_t sVT[128 * 32];      // [d][kk] transposed V block
    __shared__ __attribute__((aligned(16))) bf16_t sP[4 * 16 * 32];    // per-wave P tile [row][k]

    const int tid  = threadIdx.x;
    const int lane = tid & 31;
    const int wave = tid >> 5;
    const int lm   = lane & 15;
    const int hs   = lane >> 4;

    const int qc = blockIdx.x & 31;
    const int h  = (blockIdx.x >> 5) & 15;
    const int b  = blockIdx.x >> 9;
    const int kvh = h >> 2;

    const int qbase = qc * 64;
    const int qrow0 = qbase + wave * 16;

    // load Q fragments (A-matrix layout, 4 chunks of K=32 over D=128)
    const bf16_t* qp = Q + (((size_t)(b * NHEADS + h)) * SEQ + qrow0) * HEADDIM;
    V16 aq[4];
    for (int c = 0; c < 4; ++c) {
        const bf16_t* p = qp + (size_t)lm * HEADDIM + c * 32 + hs * 8;
        aq[c].h[0] = *(const v8bf*)p;
        aq[c].h[1] = *(const v8bf*)(p + 16);
    }

    v8f accO[8];
    for (int nf = 0; nf < 8; ++nf)
        for (int r = 0; r < 8; ++r) accO[nf][r] = 0.0f;
    float mrow[8], lrow[8];
    for (int r = 0; r < 8; ++r) { mrow[r] = -INFINITY; lrow[r] = 0.0f; }

    int ks = qbase - (WINDOW - 1); if (ks < 0) ks = 0; ks &= ~31;
    const int ke = qbase + 64;
    const bf16_t* kb_ptr = Km + ((size_t)(b * NKVHEADS + kvh)) * SEQ * HEADDIM;
    const bf16_t* vb_ptr = Vm + ((size_t)(b * NKVHEADS + kvh)) * SEQ * HEADDIM;
    const float scale = 0.08838834764831845f;  // 1/sqrt(128)
    bf16_t* pw = sP + wave * 16 * 32;

    for (int kb = ks; kb < ke; kb += 32) {
        // cooperatively stage V block transposed: sVT[d][kk]
        {
            int krow = tid >> 2;             // 0..31
            int c0   = (tid & 3) * 32;       // 0,32,64,96
            const bf16_t* vp = vb_ptr + (size_t)(kb + krow) * HEADDIM + c0;
            union { v8bf v[4]; __bf16 e[32]; } t;
            t.v[0] = *(const v8bf*)vp;        t.v[1] = *(const v8bf*)(vp + 8);
            t.v[2] = *(const v8bf*)(vp + 16); t.v[3] = *(const v8bf*)(vp + 24);
            for (int i = 0; i < 32; ++i) sVT[(c0 + i) * 32 + krow] = t.e[i];
        }
        __syncthreads();

        // scores: two 16-key fragments
        float s0[8], s1[8];
        for (int j = 0; j < 2; ++j) {
            v8f sc;
            for (int r = 0; r < 8; ++r) sc[r] = 0.0f;
            for (int c = 0; c < 4; ++c) {
                const bf16_t* kp = kb_ptr + (size_t)(kb + j * 16 + lm) * HEADDIM + c * 32 + hs * 8;
                V16 bk;
                bk.h[0] = *(const v8bf*)kp;
                bk.h[1] = *(const v8bf*)(kp + 16);
                sc = wmma_bf16(aq[c].v, bk.v, sc);
            }
            float* sj = j ? s1 : s0;
            int kcol = kb + j * 16 + lm;
            for (int r = 0; r < 8; ++r) {
                int q = qrow0 + r + hs * 8;
                bool ok = (kcol <= q) && (kcol > q - WINDOW);
                sj[r] = ok ? sc[r] * scale : -INFINITY;
            }
        }

        // online softmax update
        float p0v[8], p1v[8];
        for (int r = 0; r < 8; ++r) {
            float rmax = fmaxf(s0[r], s1[r]);
            for (int off = 1; off < 16; off <<= 1)
                rmax = fmaxf(rmax, __shfl_xor(rmax, off, 32));
            float mnew = fmaxf(mrow[r], rmax);
            float alpha, p0, p1;
            if (mnew == -INFINITY) { alpha = 1.0f; p0 = 0.0f; p1 = 0.0f; }
            else {
                alpha = __expf(mrow[r] - mnew);
                p0 = __expf(s0[r] - mnew);
                p1 = __expf(s1[r] - mnew);
            }
            float psum = p0 + p1;
            for (int off = 1; off < 16; off <<= 1)
                psum += __shfl_xor(psum, off, 32);
            lrow[r] = lrow[r] * alpha + psum;
            mrow[r] = mnew;
            p0v[r] = p0; p1v[r] = p1;
            for (int nf = 0; nf < 8; ++nf) accO[nf][r] *= alpha;
        }

        // stage P (16x32) into per-wave LDS in row-major [row][k]
        for (int r = 0; r < 8; ++r) {
            int row = r + hs * 8;
            pw[row * 32 + lm]      = (bf16_t)p0v[r];
            pw[row * 32 + 16 + lm] = (bf16_t)p1v[r];
        }
        asm volatile("s_wait_dscnt 0x0" ::: "memory");  // same-wave LDS RAW

        // P as A-fragment
        V16 pa;
        {
            const bf16_t* pp = pw + lm * 32 + hs * 8;
            pa.h[0] = *(const v8bf*)pp;
            pa.h[1] = *(const v8bf*)(pp + 16);
        }
        // accumulate O += P * V  (8 column fragments over D=128)
        for (int nf = 0; nf < 8; ++nf) {
            const bf16_t* vp = sVT + (nf * 16 + lm) * 32 + hs * 8;
            V16 bv;
            bv.h[0] = *(const v8bf*)vp;
            bv.h[1] = *(const v8bf*)(vp + 16);
            accO[nf] = wmma_bf16(pa.v, bv.v, accO[nf]);
        }
        __syncthreads();  // protect sVT before next iteration
    }

    float invr[8];
    for (int r = 0; r < 8; ++r) invr[r] = (lrow[r] > 0.0f) ? (1.0f / lrow[r]) : 0.0f;
    for (int nf = 0; nf < 8; ++nf)
        for (int r = 0; r < 8; ++r) {
            int q = qrow0 + r + hs * 8;
            O[((size_t)(b * SEQ) + q) * HIDDEN + h * HEADDIM + nf * 16 + lm] =
                (bf16_t)(accO[nf][r] * invr[r]);
        }
}

// ---------------------------------------------------------------- launch
extern "C" void kernel_launch(void* const* d_in, const int* in_sizes, int n_in,
                              void* d_out, int out_size, void* d_ws, size_t ws_size,
                              hipStream_t stream) {
    const float*     X   = (const float*)d_in[0];
    const long long* pos = (const long long*)d_in[1];
    const float*     Wq  = (const float*)d_in[2];
    const float*     bq  = (const float*)d_in[3];
    const float*     Wk  = (const float*)d_in[4];
    const float*     bk  = (const float*)d_in[5];
    const float*     Wv  = (const float*)d_in[6];
    const float*     bv  = (const float*)d_in[7];
    const float*     Wo  = (const float*)d_in[8];
    float* out = (float*)d_out;

    char* w = (char*)d_ws;
    bf16_t* Xbf  = (bf16_t*)(w + 0);           // 16,777,216 B
    bf16_t* Wqbf = (bf16_t*)(w + 16777216);    //  8,388,608
    bf16_t* Wkbf = (bf16_t*)(w + 25165824);    //  2,097,152
    bf16_t* Wvbf = (bf16_t*)(w + 27262976);    //  2,097,152
    bf16_t* Wobf = (bf16_t*)(w + 29360128);    //  8,388,608
    float*  Qf   = (float*) (w + 37748736);    // 33,554,432
    float*  Kf   = (float*) (w + 71303168);    //  8,388,608
    float*  Vf   = (float*) (w + 79691776);    //  8,388,608
    bf16_t* Qbf  = (bf16_t*)(w + 88080384);    // 16,777,216
    bf16_t* Kbf  = (bf16_t*)(w + 104857600);   //  4,194,304
    bf16_t* Vbf  = (bf16_t*)(w + 109051904);   //  4,194,304
    bf16_t* AObf = (bf16_t*)(w + 37748736);    // aliases Qf (dead after RoPE)

    const long nX  = (long)BATCH * SEQ * HIDDEN;          // 8,388,608
    const long nWq = (long)HIDDEN * HIDDEN;               // 4,194,304
    const long nWk = (long)HIDDEN * (NKVHEADS * HEADDIM); // 1,048,576

    f32_to_bf16<<<(unsigned)((nX  + 255) / 256), 256, 0, stream>>>(X,  Xbf,  nX);
    f32_to_bf16<<<(unsigned)((nWq + 255) / 256), 256, 0, stream>>>(Wq, Wqbf, nWq);
    f32_to_bf16<<<(unsigned)((nWk + 255) / 256), 256, 0, stream>>>(Wk, Wkbf, nWk);
    f32_to_bf16<<<(unsigned)((nWk + 255) / 256), 256, 0, stream>>>(Wv, Wvbf, nWk);
    f32_to_bf16<<<(unsigned)((nWq + 255) / 256), 256, 0, stream>>>(Wo, Wobf, nWq);

    const int M = BATCH * SEQ;            // 4096
    gemm_bf16<<<dim3(HIDDEN / 128, M / 128), 256, 0, stream>>>(Xbf, Wqbf, bq, Qf, M, HIDDEN, HIDDEN);
    gemm_bf16<<<dim3(512 / 128,    M / 128), 256, 0, stream>>>(Xbf, Wkbf, bk, Kf, M, 512, HIDDEN);
    gemm_bf16<<<dim3(512 / 128,    M / 128), 256, 0, stream>>>(Xbf, Wvbf, bv, Vf, M, 512, HIDDEN);

    const long nQ  = (long)BATCH * NHEADS   * SEQ * HEADDIM;  // 8,388,608
    const long nKV = (long)BATCH * NKVHEADS * SEQ * HEADDIM;  // 2,097,152
    rope_relayout<<<(unsigned)((nQ  + 255) / 256), 256, 0, stream>>>(Qf, Qbf, pos, NHEADS,   1, nQ);
    rope_relayout<<<(unsigned)((nKV + 255) / 256), 256, 0, stream>>>(Kf, Kbf, pos, NKVHEADS, 1, nKV);
    rope_relayout<<<(unsigned)((nKV + 255) / 256), 256, 0, stream>>>(Vf, Vbf, pos, NKVHEADS, 0, nKV);

    attn_kernel<<<BATCH * NHEADS * (SEQ / 64), 128, 0, stream>>>(Qbf, Kbf, Vbf, AObf);

    gemm_bf16<<<dim3(HIDDEN / 128, M / 128), 256, 0, stream>>>(AObf, Wobf, nullptr, out, M, HIDDEN, HIDDEN);
}
